// ScaledDotProductAttention_21466246545789
// MI455X (gfx1250) — matrix operand order repaired and checked by reading, not compiled, not run on hardware
//
#include <hip/hip_runtime.h>

typedef __attribute__((ext_vector_type(16))) __bf16 v16bf;
typedef __attribute__((ext_vector_type(8)))  __bf16 bf16x8;
typedef __attribute__((ext_vector_type(8)))  float  v8f;
typedef int v4i __attribute__((vector_size(16)));

constexpr int kH  = 768;
constexpr int kNH = 12;
constexpr int kHS = 64;
constexpr int kT  = 512;
constexpr int kB  = 16;
constexpr int kM  = kB * kT;          // 8192 rows
constexpr int BM  = 128, BN = 256, BK = 32;
constexpr int LDA = 40;               // padded LDS row stride (elements) -> 80B, bank-friendly
constexpr int LDB = 40;

static __device__ __forceinline__ v16bf combine16(bf16x8 lo, bf16x8 hi) {
  v16bf r;
#pragma unroll
  for (int i = 0; i < 8; ++i) { r[i] = lo[i]; r[i + 8] = hi[i]; }
  return r;
}

// ---- 16-byte global->LDS copy: CDNA5 async path if available, else staged ----
#if __has_builtin(__builtin_amdgcn_global_load_async_to_lds_b128)
#define ASYNC_CP 1
static __device__ __forceinline__ void cp16(const __bf16* g, __bf16* l) {
  __builtin_amdgcn_global_load_async_to_lds_b128(
      (__attribute__((address_space(1))) v4i*)g,
      (__attribute__((address_space(3))) v4i*)l, 0, 0);
}
static __device__ __forceinline__ void cp_wait() {
#if __has_builtin(__builtin_amdgcn_s_wait_asynccnt)
  __builtin_amdgcn_s_wait_asynccnt(0);
#else
  asm volatile("s_wait_asynccnt 0" ::: "memory");
#endif
}
#else
#define ASYNC_CP 0
static __device__ __forceinline__ void cp16(const __bf16* g, __bf16* l) {
  *(bf16x8*)l = *(const bf16x8*)g;
}
static __device__ __forceinline__ void cp_wait() {}
#endif

// ---------------------------------------------------------------- converters
__global__ __launch_bounds__(256) void k_f32_to_bf16(const float* __restrict__ in,
                                                     __bf16* __restrict__ out, int n) {
  int i = (blockIdx.x * 256 + threadIdx.x) * 8;
  if (i + 7 < n) {
    float4 a = *(const float4*)(in + i);
    float4 b = *(const float4*)(in + i + 4);
    bf16x8 o;
    o[0] = (__bf16)a.x; o[1] = (__bf16)a.y; o[2] = (__bf16)a.z; o[3] = (__bf16)a.w;
    o[4] = (__bf16)b.x; o[5] = (__bf16)b.y; o[6] = (__bf16)b.z; o[7] = (__bf16)b.w;
    *(bf16x8*)(out + i) = o;
  }
}

// in: [K][N] f32 row-major -> out: [N][K] bf16 (transposed). grid = (K, N/256).
__global__ __launch_bounds__(256) void k_transpose_bf16(const float* __restrict__ in,
                                                        __bf16* __restrict__ out,
                                                        int K, int N) {
  int k = blockIdx.x;
  int n = blockIdx.y * 256 + threadIdx.x;
  if (n < N) out[(size_t)n * K + k] = (__bf16)in[(size_t)k * N + n];
}

// ---------------------------------------------------------------- GEMM (bf16 WMMA)
// C[M][N] = A[M][K] * Bt[N][K]^T + bias.  mode 0: fp32 out.  mode 1: QKV scatter.
// 128x256 block tile, 8 waves, 64x64 per wave -> 16 WMMA per 32-K step.
__global__ __launch_bounds__(256) void gemm_bf16_wmma(
    const __bf16* __restrict__ A, const __bf16* __restrict__ Bt,
    const float* __restrict__ bias, int M, int N, int K, int mode,
    float* __restrict__ outF,
    __bf16* __restrict__ qb, __bf16* __restrict__ kbuf, __bf16* __restrict__ vt)
{
  __shared__ alignas(64) __bf16 sA[2][BM * LDA];
  __shared__ alignas(64) __bf16 sB[2][BN * LDB];

  const int tid = threadIdx.x;
  const int m0 = blockIdx.y * BM;
  const int n0 = blockIdx.x * BN;
  const int w = tid >> 5, lane = tid & 31, lane16 = lane & 15, hf = lane >> 4;
  const int wm = (w & 1) * 64;        // 2 waves along M
  const int wn = (w >> 1) * 64;       // 4 waves along N

  // staging coordinates (per thread): A: 2x16B, B: 4x16B per K-step
  const int arow = tid >> 1, acol = (tid & 1) * 16;   // A: 128 rows x 32
  const int brow = tid;                               // B: 256 rows x 32

  v8f acc[4][4];
#pragma unroll
  for (int i = 0; i < 4; ++i)
#pragma unroll
    for (int j = 0; j < 4; ++j) acc[i][j] = (v8f){};

  const int nsteps = K / BK;
  auto stage = [&](int buf, int k0) {
    const __bf16* ga = A + (size_t)(m0 + arow) * K + k0 + acol;
    __bf16* la = &sA[buf][arow * LDA + acol];
    cp16(ga, la);
    cp16(ga + 8, la + 8);
    const __bf16* gb = Bt + (size_t)(n0 + brow) * K + k0;
    __bf16* lb = &sB[buf][brow * LDB];
#pragma unroll
    for (int c = 0; c < 4; ++c) cp16(gb + c * 8, lb + c * 8);
  };

  stage(0, 0);
  cp_wait();
  __syncthreads();

  for (int ks = 0; ks < nsteps; ++ks) {
    const int cur = ks & 1, nxt = cur ^ 1;
    if (ks + 1 < nsteps) stage(nxt, (ks + 1) * BK);

    v16bf af[4], bfr[4];
#pragma unroll
    for (int mt = 0; mt < 4; ++mt) {
      const __bf16* p = &sA[cur][(wm + mt * 16 + lane16) * LDA];
      af[mt] = combine16(*(const bf16x8*)(p + hf * 16),
                         *(const bf16x8*)(p + hf * 16 + 8));
    }
#pragma unroll
    for (int nt = 0; nt < 4; ++nt) {
      const __bf16* p = &sB[cur][(wn + nt * 16 + lane16) * LDB + hf * 16];
      bfr[nt] = combine16(*(const bf16x8*)(p), *(const bf16x8*)(p + 8));
    }
#pragma unroll
    for (int mt = 0; mt < 4; ++mt)
#pragma unroll
      for (int nt = 0; nt < 4; ++nt)
        acc[mt][nt] = __builtin_amdgcn_wmma_f32_16x16x32_bf16(
            false, af[mt], false, bfr[nt], (short)0, acc[mt][nt], false, false);

    if (ks + 1 < nsteps) cp_wait();
    __syncthreads();
  }

  // epilogue: D layout row m = r + 8*hf, col n = lane16 (no divisions)
#pragma unroll
  for (int nt = 0; nt < 4; ++nt) {
    const int gn = n0 + wn + nt * 16 + lane16;
    const float bv = bias[gn];
    const int which = (gn >= 2 * kH) ? 2 : (gn >= kH ? 1 : 0);
    const int c = gn - which * kH;
    const int hh = c >> 6, d = c & (kHS - 1);
#pragma unroll
    for (int mt = 0; mt < 4; ++mt) {
#pragma unroll
      for (int r = 0; r < 8; ++r) {
        const int gm = m0 + wm + mt * 16 + r + hf * 8;
        const float v = acc[mt][nt][r] + bv;
        if (mode == 0) {
          outF[(size_t)gm * N + gn] = v;
        } else {
          const int bb = gm >> 9, t = gm & (kT - 1);
          if (which == 0)       // Q, pre-scaled by 1/sqrt(HS)
            qb[((size_t)(bb * kNH + hh) * kT + t) * kHS + d] = (__bf16)(v * 0.125f);
          else if (which == 1)  // K
            kbuf[((size_t)(bb * kNH + hh) * kT + t) * kHS + d] = (__bf16)v;
          else                  // V transposed: [B,NH,HS,T]
            vt[((size_t)(bb * kNH + hh) * kHS + d) * kT + t] = (__bf16)v;
        }
      }
    }
  }
}

// ---------------------------------------------------------------- attention
__global__ __launch_bounds__(256) void attn_wmma(
    const __bf16* __restrict__ qb, const __bf16* __restrict__ kb,
    const __bf16* __restrict__ vt, const long long* __restrict__ amask,
    __bf16* __restrict__ ybf)
{
  __shared__ alignas(64) __bf16 pbuf[8 * 16 * 40];   // per-wave 16x32 P tile, padded rows

  const int tid = threadIdx.x;
  const int w = tid >> 5, lane = tid & 31, lane16 = lane & 15, hf = lane >> 4;
  const int bh = blockIdx.x >> 2, qt = blockIdx.x & 3;
  const int b = bh / kNH, h = bh % kNH;
  const int qr0 = qt * 128 + w * 16;

  // Q fragments (A layout), kept in registers for the whole key sweep
  const __bf16* qp = qb + ((size_t)bh * kT + qr0 + lane16) * kHS;
  v16bf aq[2];
#pragma unroll
  for (int c = 0; c < 2; ++c)
    aq[c] = combine16(*(const bf16x8*)(qp + c * 32 + hf * 16),
                      *(const bf16x8*)(qp + c * 32 + hf * 16 + 8));

  v8f o[4];
#pragma unroll
  for (int i = 0; i < 4; ++i) o[i] = (v8f){};
  float rmax[8], rsum[8];
#pragma unroll
  for (int r = 0; r < 8; ++r) { rmax[r] = -3.0e38f; rsum[r] = 0.f; }

  __bf16* pw = pbuf + w * (16 * 40);
  const float NEG = -3.0e38f;

  for (int kt = 0; kt < kT / 32; ++kt) {
    const int kbase = kt * 32;
    const bool f0 = amask[(size_t)b * kT + kbase + lane16] != 0;
    const bool f1 = amask[(size_t)b * kT + kbase + 16 + lane16] != 0;

    // scores: S[16q x 32k] as two 16x16 f32 accumulators
    v8f s[2];
#pragma unroll
    for (int js = 0; js < 2; ++js) {
      const __bf16* kp = kb + ((size_t)bh * kT + kbase + js * 16 + lane16) * kHS;
      v16bf bk0 = *(const v16bf*)(kp + hf * 16);        // d = 0..31 chunk
      v16bf bk1 = *(const v16bf*)(kp + 32 + hf * 16);   // d = 32..63 chunk
      v8f z = (v8f){};
      z = __builtin_amdgcn_wmma_f32_16x16x32_bf16(false, aq[0], false, bk0, (short)0, z, false, false);
      z = __builtin_amdgcn_wmma_f32_16x16x32_bf16(false, aq[1], false, bk1, (short)0, z, false, false);
      s[js] = z;
    }
#pragma unroll
    for (int r = 0; r < 8; ++r) {
      if (!f0) s[0][r] = NEG;
      if (!f1) s[1][r] = NEG;
    }

    // online softmax; row m = r + 8*hf lives in 16 lanes of this half
    float p0[8], p1[8];
#pragma unroll
    for (int r = 0; r < 8; ++r) {
      float rm = fmaxf(s[0][r], s[1][r]);
      rm = fmaxf(rm, __shfl_xor(rm, 1, 32));
      rm = fmaxf(rm, __shfl_xor(rm, 2, 32));
      rm = fmaxf(rm, __shfl_xor(rm, 4, 32));
      rm = fmaxf(rm, __shfl_xor(rm, 8, 32));
      float nm  = fmaxf(rmax[r], rm);
      float fac = __expf(rmax[r] - nm);
      rmax[r] = nm;
#pragma unroll
      for (int dn = 0; dn < 4; ++dn) o[dn][r] *= fac;
      float e0 = __expf(s[0][r] - nm);
      float e1 = __expf(s[1][r] - nm);
      float ps = e0 + e1;
      ps += __shfl_xor(ps, 1, 32);
      ps += __shfl_xor(ps, 2, 32);
      ps += __shfl_xor(ps, 4, 32);
      ps += __shfl_xor(ps, 8, 32);
      rsum[r] = rsum[r] * fac + ps;
      p0[r] = e0; p1[r] = e1;
    }

    // transpose P (D layout) -> A layout through per-wave LDS (padded stride)
#pragma unroll
    for (int r = 0; r < 8; ++r) {
      int m = r + hf * 8;
      pw[m * 40 + lane16]      = (__bf16)p0[r];
      pw[m * 40 + 16 + lane16] = (__bf16)p1[r];
    }
    __builtin_amdgcn_wave_barrier();
    const __bf16* pr = pw + lane16 * 40;
    v16bf ap = combine16(*(const bf16x8*)(pr + hf * 16),
                         *(const bf16x8*)(pr + hf * 16 + 8));

    // O += P @ V  (V stored [d][t]: B-fragment is contiguous along keys)
#pragma unroll
    for (int dn = 0; dn < 4; ++dn) {
      const __bf16* vp = vt + ((size_t)bh * kHS + dn * 16 + lane16) * kT + kbase + hf * 16;
      v16bf bv = *(const v16bf*)vp;
      o[dn] = __builtin_amdgcn_wmma_f32_16x16x32_bf16(false, ap, false, bv, (short)0, o[dn], false, false);
    }
    __builtin_amdgcn_wave_barrier();
  }

  // normalize and write y in [B,T,H] bf16 for the projection GEMM
#pragma unroll
  for (int r = 0; r < 8; ++r) {
    float inv = 1.0f / rsum[r];
    int gm = qr0 + r + hf * 8;
#pragma unroll
    for (int dn = 0; dn < 4; ++dn) {
      int gc = h * kHS + dn * 16 + lane16;
      ybf[((size_t)b * kT + gm) * kH + gc] = (__bf16)(o[dn][r] * inv);
    }
  }
}

// ---------------------------------------------------------------- launcher
extern "C" void kernel_launch(void* const* d_in, const int* in_sizes, int n_in,
                              void* d_out, int out_size, void* d_ws, size_t ws_size,
                              hipStream_t stream) {
  const float*     x     = (const float*)d_in[0];
  const long long* amask = (const long long*)d_in[1];
  const float*     Wqkv  = (const float*)d_in[2];
  const float*     bqkv  = (const float*)d_in[3];
  const float*     Wproj = (const float*)d_in[4];
  const float*     bproj = (const float*)d_in[5];
  float*           out   = (float*)d_out;

  char* ws = (char*)d_ws;
  size_t off = 0;
  auto alloc = [&](size_t bytes) {
    char* p = ws + off;
    off += (bytes + 255) & ~(size_t)255;
    return p;
  };
  __bf16* Xbf    = (__bf16*)alloc((size_t)kM * kH * 2);
  __bf16* WqkvT  = (__bf16*)alloc((size_t)3 * kH * kH * 2);   // [2304][768]
  __bf16* WprojT = (__bf16*)alloc((size_t)kH * kH * 2);       // [768][768]
  __bf16* qb     = (__bf16*)alloc((size_t)kB * kNH * kT * kHS * 2);
  __bf16* kb     = (__bf16*)alloc((size_t)kB * kNH * kT * kHS * 2);
  __bf16* vt     = (__bf16*)alloc((size_t)kB * kNH * kHS * kT * 2);
  __bf16* ybf    = (__bf16*)alloc((size_t)kM * kH * 2);
  (void)ws_size; (void)in_sizes; (void)n_in; (void)out_size;

  // 1) fp32 -> bf16 conversions (weights transposed to [N][K])
  const int nX = kM * kH;
  k_f32_to_bf16<<<nX / (256 * 8), 256, 0, stream>>>(x, Xbf, nX);
  k_transpose_bf16<<<dim3(kH, (3 * kH) / 256), 256, 0, stream>>>(Wqkv, WqkvT, kH, 3 * kH);
  k_transpose_bf16<<<dim3(kH, kH / 256), 256, 0, stream>>>(Wproj, WprojT, kH, kH);

  // 2) QKV projection + bias + per-head scatter (Q pre-scaled by 1/8)
  gemm_bf16_wmma<<<dim3((3 * kH) / BN, kM / BM), 256, 0, stream>>>(
      Xbf, WqkvT, bqkv, kM, 3 * kH, kH, 1, nullptr, qb, kb, vt);

  // 3) flash-style attention, WMMA scores + WMMA PV
  attn_wmma<<<dim3(kB * kNH * (kT / 128)), 256, 0, stream>>>(qb, kb, vt, amask, ybf);

  // 4) output projection + bias -> fp32 result
  gemm_bf16_wmma<<<dim3(kH / BN, kM / BM), 256, 0, stream>>>(
      ybf, WprojT, bproj, kM, kH, kH, 0, out, nullptr, nullptr, nullptr);
}
